// NemotronHMamba2Mixer_21723944583766
// MI455X (gfx1250) — matrix operand order repaired and checked by reading, not compiled, not run on hardware
//
#include <hip/hip_runtime.h>
#include <hip/hip_bf16.h>

// ---------------------------------------------------------------------------
// NemotronH Mamba2 mixer for MI455X (gfx1250, wave32, WMMA).
// All heavy math runs through v_wmma_f32_16x16x32_bf16; tile staging in the
// SSM kernels uses CDNA5 async global->LDS copies (ASYNCcnt).
// Workspace requirement: ~1.18 GB.
// ---------------------------------------------------------------------------

constexpr int cB    = 2;
constexpr int cL    = 2048;
constexpr int cBL   = cB * cL;          // 4096 rows
constexpr int cHID  = 4096;
constexpr int cH    = 128;              // heads
constexpr int cP    = 64;               // head dim
constexpr int cN    = 128;              // state dim
constexpr int cG    = 8;                // groups
constexpr int cK    = 4;                // conv width
constexpr int cCH   = 128;              // chunk length
constexpr int cNC   = cL / cCH;         // 16 chunks
constexpr int cINTER = cH * cP;         // 8192
constexpr int cGN   = cG * cN;          // 1024
constexpr int cCONVD = cINTER + 2 * cGN;        // 10240
constexpr int cPROJ  = cINTER + cCONVD + cH;    // 18560
constexpr int POFF_HBC = cINTER;                // 8192
constexpr int POFF_DT  = cINTER + cCONVD;       // 18432
constexpr float cEPS = 1e-5f;

typedef __bf16 v16bf __attribute__((ext_vector_type(16)));
typedef __bf16 v8bf  __attribute__((ext_vector_type(8)));
typedef float  v8f   __attribute__((ext_vector_type(8)));

// A-matrix fragment (16x32 bf16, M = lane&15): lane-half h holds K = k0+16h .. k0+16h+15.
__device__ __forceinline__ v16bf frag_a(const __bf16* row, int k0, int half) {
  return *(const v16bf*)(row + k0 + half * 16);
}
// B-matrix fragment (32x16 bf16, N = lane&15): lane-half h holds K = k0+16h .. k0+16h+15.
__device__ __forceinline__ v16bf frag_b(const __bf16* row, int k0, int half) {
  return *(const v16bf*)(row + k0 + half * 16);
}
__device__ __forceinline__ v8f wmma_bf16(v16bf a, v16bf b, v8f c) {
  return __builtin_amdgcn_wmma_f32_16x16x32_bf16(false, a, false, b,
                                                 (short)0, c, false, false);
}

// CDNA5 async global -> LDS copy, 16 bytes per lane (ASYNCcnt tracked).
// Generic LDS pointers carry the LDS byte offset in their low 32 bits.
__device__ __forceinline__ void async_ld16(void* lds, const void* g) {
  uint32_t loff = (uint32_t)(uintptr_t)lds;
  asm volatile("global_load_async_to_lds_b128 %0, %1, off"
               :: "v"(loff), "v"(g) : "memory");
}
__device__ __forceinline__ void wait_async0() {
  asm volatile("s_wait_asynccnt 0x0" ::: "memory");
}

__device__ __forceinline__ float softplus_f(float x) {
  return (x > 20.f) ? x : log1pf(__expf(x));
}
__device__ __forceinline__ float silu_f(float x) {
  return x / (1.f + __expf(-x));
}

// ---------------------------------------------------------------------------
// fp32 -> bf16 cast (grid stride)
// ---------------------------------------------------------------------------
__global__ void k_cast_bf16(const float* __restrict__ src,
                            __bf16* __restrict__ dst, size_t n) {
  size_t i = blockIdx.x * (size_t)blockDim.x + threadIdx.x;
  size_t stride = (size_t)gridDim.x * blockDim.x;
  for (; i < n; i += stride) dst[i] = (__bf16)src[i];
}

// ---------------------------------------------------------------------------
// C[M,N] (f32) = A[M,K] (bf16, row major) @ Bw[N,K]^T (bf16, row major)
// block = 256 threads = 8 waves; block tile 128(M) x 128(N); wave tile 32x64.
// 8 WMMAs per 12 b128 loads; weights stream through L2 (192 MB).
// ---------------------------------------------------------------------------
__global__ void __launch_bounds__(256)
k_gemm_nt_bf16(const __bf16* __restrict__ A, const __bf16* __restrict__ Bw,
               float* __restrict__ C, int M, int N, int K) {
  const int tid  = threadIdx.x;
  const int wid  = tid >> 5;
  const int lane = tid & 31;
  const int half = lane >> 4;
  const int lr   = lane & 15;
  const int wm = blockIdx.y * 128 + (wid & 3) * 32;
  const int wn = blockIdx.x * 128 + (wid >> 2) * 64;

  const __bf16* a0 = A + (size_t)(wm + lr) * K;
  const __bf16* a1 = A + (size_t)(wm + 16 + lr) * K;
  const __bf16* b0 = Bw + (size_t)(wn + lr) * K;
  const __bf16* b1 = Bw + (size_t)(wn + 16 + lr) * K;
  const __bf16* b2 = Bw + (size_t)(wn + 32 + lr) * K;
  const __bf16* b3 = Bw + (size_t)(wn + 48 + lr) * K;

  v8f acc[2][4];
#pragma unroll
  for (int i = 0; i < 2; ++i)
#pragma unroll
    for (int j = 0; j < 4; ++j) acc[i][j] = (v8f){};

  for (int k0 = 0; k0 < K; k0 += 32) {
    __builtin_prefetch(a0 + k0 + 256, 0, 3);   // global_prefetch_b8
    __builtin_prefetch(b0 + k0 + 256, 0, 3);
    __builtin_prefetch(b2 + k0 + 256, 0, 3);
    v16bf fa0 = frag_a(a0, k0, half);
    v16bf fa1 = frag_a(a1, k0, half);
    v16bf fb0 = frag_b(b0, k0, half);
    v16bf fb1 = frag_b(b1, k0, half);
    v16bf fb2 = frag_b(b2, k0, half);
    v16bf fb3 = frag_b(b3, k0, half);
    acc[0][0] = wmma_bf16(fa0, fb0, acc[0][0]);
    acc[0][1] = wmma_bf16(fa0, fb1, acc[0][1]);
    acc[0][2] = wmma_bf16(fa0, fb2, acc[0][2]);
    acc[0][3] = wmma_bf16(fa0, fb3, acc[0][3]);
    acc[1][0] = wmma_bf16(fa1, fb0, acc[1][0]);
    acc[1][1] = wmma_bf16(fa1, fb1, acc[1][1]);
    acc[1][2] = wmma_bf16(fa1, fb2, acc[1][2]);
    acc[1][3] = wmma_bf16(fa1, fb3, acc[1][3]);
  }
#pragma unroll
  for (int i = 0; i < 2; ++i)
#pragma unroll
    for (int j = 0; j < 4; ++j)
#pragma unroll
      for (int r = 0; r < 8; ++r) {
        const int row = wm + i * 16 + r + 8 * half;
        const int col = wn + j * 16 + lr;
        C[(size_t)row * N + col] = acc[i][j][r];
      }
}

// ---------------------------------------------------------------------------
// Causal depthwise conv(K=4) + bias + SiLU over hBC columns of proj.
// Splits into hs (bf16), X = hs*softplus(dt) (bf16), B (bf16), C (bf16).
// ---------------------------------------------------------------------------
__global__ void k_conv_silu(const float* __restrict__ proj,
                            const float* __restrict__ cw,
                            const float* __restrict__ cb,
                            const float* __restrict__ dt_bias,
                            __bf16* __restrict__ hsb, __bf16* __restrict__ xdt,
                            __bf16* __restrict__ bbf, __bf16* __restrict__ cbf) {
  size_t idx = blockIdx.x * (size_t)blockDim.x + threadIdx.x;
  if (idx >= (size_t)cBL * cCONVD) return;
  const int ch = (int)(idx % cCONVD);
  const int row = (int)(idx / cCONVD);
  const int b = row / cL, l = row % cL;

  float s = cb[ch];
#pragma unroll
  for (int k = 0; k < cK; ++k) {
    const int ls = l - (cK - 1) + k;
    if (ls >= 0)
      s += cw[ch * cK + k] * proj[(size_t)(b * cL + ls) * cPROJ + POFF_HBC + ch];
  }
  s = silu_f(s);

  if (ch < cINTER) {
    const int h = ch >> 6;  // P = 64
    const float dt = softplus_f(proj[(size_t)row * cPROJ + POFF_DT + h] + dt_bias[h]);
    hsb[(size_t)row * cINTER + ch] = (__bf16)s;
    xdt[(size_t)row * cINTER + ch] = (__bf16)(s * dt);
  } else if (ch < cINTER + cGN) {
    bbf[(size_t)row * cGN + (ch - cINTER)] = (__bf16)s;
  } else {
    cbf[(size_t)row * cGN + (ch - cINTER - cGN)] = (__bf16)s;
  }
}

// ---------------------------------------------------------------------------
// Per-(b, chunk, head) intra-chunk pipeline, all GEMMs via bf16 WMMA.
//  0. issue async global->LDS copies of C(l,n), B(s,n) tiles (overlap w/ scan)
//  1. a_dt = A*dt, inclusive scan -> s_acs (also written to acs_g)
//  2. derive Bdt(n,l)=B*decay (from LDS B) and Xt(p,l)
//  3. Gm = C @ B^T, mask * exp(acs[l]-acs[s]) -> sM(l,s) bf16
//  4. Y_diag = sM @ X  -> yout
//  5. states S(p,n) = Xt @ Bdt^T -> states
// Dynamic LDS: 73728 bf16 = 144 KB (WGP has 320 KB).
// ---------------------------------------------------------------------------
__global__ void __launch_bounds__(256)
k_chunk(const float* __restrict__ proj, const __bf16* __restrict__ xdt,
        const __bf16* __restrict__ bbf, const __bf16* __restrict__ cbf,
        const float* __restrict__ dt_bias, const float* __restrict__ A_log,
        float* __restrict__ yout, float* __restrict__ states,
        float* __restrict__ acs_g) {
  extern __shared__ __bf16 pool[];
  __bf16* sC   = pool;            // [128][128] (l,n)
  __bf16* sB   = pool + 16384;    // [128][128] (s,n)
  __bf16* sM   = pool + 32768;    // [128][128] (l,s)
  __bf16* sXt  = pool + 49152;    // [64][128]  (p,l)
  __bf16* sBdt = pool + 57344;    // [128][128] (n,l)
  __shared__ float s_acs[cCH];

  const int tid  = threadIdx.x;
  const int wid  = tid >> 5;
  const int lane = tid & 31;
  const int half = lane >> 4;
  const int lr   = lane & 15;

  const int idx = blockIdx.x;
  const int h = idx % cH;
  const int c = (idx / cH) % cNC;
  const int b = idx / (cH * cNC);
  const int g = h % cG;  // jnp.tile: head h -> group h % G
  const int base = b * cL + c * cCH;

  // --- 0. async DMA of the C and B tiles into LDS (16B per lane) ---
  for (int ch16 = tid; ch16 < 2048; ch16 += 256) {
    const int l = ch16 >> 4;             // 16 x 16B chunks per 128-elem row
    const int n8 = (ch16 & 15) * 8;
    const size_t goff = (size_t)(base + l) * cGN + g * cN + n8;
    async_ld16(sC + ch16 * 8, cbf + goff);
    async_ld16(sB + ch16 * 8, bbf + goff);
  }

  // --- 1. A*dt and inclusive scan (overlaps with async DMA) ---
  if (tid < cCH) {
    const float x = proj[(size_t)(base + tid) * cPROJ + POFF_DT + h] + dt_bias[h];
    s_acs[tid] = -__expf(A_log[h]) * softplus_f(x);
  }
  __syncthreads();
  for (int offd = 1; offd < cCH; offd <<= 1) {
    float v = 0.f;
    const bool act = (tid < cCH) && (tid >= offd);
    if (act) v = s_acs[tid - offd];
    __syncthreads();
    if (act) s_acs[tid] += v;
    __syncthreads();
  }
  if (tid < cCH) acs_g[(size_t)idx * cCH + tid] = s_acs[tid];

  wait_async0();
  __syncthreads();
  const float a_last = s_acs[cCH - 1];

  // --- 2. derived tiles: Bdt (transposed, decayed) and Xt (transposed) ---
  for (int i = tid; i < cCH * cN; i += 256) {
    const int l = i >> 7, n = i & 127;
    sBdt[n * cCH + l] = (__bf16)((float)sB[i] * __expf(a_last - s_acs[l]));
  }
  for (int i = tid; i < cP * cCH; i += 256) {
    const int p = i >> 7, l = i & 127;
    sXt[i] = xdt[(size_t)(base + l) * cINTER + h * cP + p];
  }
  __syncthreads();

  // --- 3. Gm = C @ B^T with causal decay mask -> sM (bf16) ---
  {
    const int lt = wid;  // each wave owns one 16-row l-tile (all 8 s-tiles)
    v8f acc[8] = {};
    const __bf16* arow = sC + (lt * 16 + lr) * cN;
    for (int k0 = 0; k0 < cN; k0 += 32) {
      v16bf fa = frag_a(arow, k0, half);
      for (int st = 0; st <= lt; ++st) {
        v16bf fb = frag_b(sB + (st * 16 + lr) * cN, k0, half);
        acc[st] = wmma_bf16(fa, fb, acc[st]);
      }
    }
    for (int st = 0; st < 8; ++st) {
#pragma unroll
      for (int r = 0; r < 8; ++r) {
        const int li = lt * 16 + r + 8 * half;
        const int si = st * 16 + lr;
        float m = 0.f;
        if (st <= lt && si <= li) m = acc[st][r] * __expf(s_acs[li] - s_acs[si]);
        sM[li * cCH + si] = (__bf16)m;
      }
    }
  }
  __syncthreads();

  // --- 4. Y_diag = M @ X (128 x 64) ---
  {
    const int lt = wid;
    v8f acc[4] = {};
    const __bf16* arow = sM + (lt * 16 + lr) * cCH;
    for (int k0 = 0; k0 < cCH; k0 += 32) {
      v16bf fa = frag_a(arow, k0, half);
#pragma unroll
      for (int pt = 0; pt < 4; ++pt) {
        v16bf fb = frag_b(sXt + (pt * 16 + lr) * cCH, k0, half);
        acc[pt] = wmma_bf16(fa, fb, acc[pt]);
      }
    }
#pragma unroll
    for (int pt = 0; pt < 4; ++pt)
#pragma unroll
      for (int r = 0; r < 8; ++r) {
        const int li = lt * 16 + r + 8 * half;
        const int p = pt * 16 + lr;
        yout[(size_t)(base + li) * cINTER + h * cP + p] = acc[pt][r];
      }
  }

  // --- 5. states S[p][n] = sum_l X[l][p] * Bdecay[l][n] ---
  {
    const int pt = wid & 3;
    const int ntg = (wid >> 2) * 4;
    v8f acc[4] = {};
    const __bf16* arow = sXt + (pt * 16 + lr) * cCH;
    for (int k0 = 0; k0 < cCH; k0 += 32) {
      v16bf fa = frag_a(arow, k0, half);
#pragma unroll
      for (int j = 0; j < 4; ++j) {
        v16bf fb = frag_b(sBdt + ((ntg + j) * 16 + lr) * cCH, k0, half);
        acc[j] = wmma_bf16(fa, fb, acc[j]);
      }
    }
    float* sout = states + (size_t)idx * (cP * cN);
#pragma unroll
    for (int j = 0; j < 4; ++j)
#pragma unroll
      for (int r = 0; r < 8; ++r) {
        const int p = pt * 16 + r + 8 * half;
        const int n = (ntg + j) * 16 + lr;
        sout[p * cN + n] = acc[j][r];
      }
  }
}

// ---------------------------------------------------------------------------
// Inter-chunk recurrence per (b, h): running = running*exp(asum_c) + S_c,
// with prev_states[c] = running before update.
// ---------------------------------------------------------------------------
__global__ void __launch_bounds__(256)
k_interchunk(const float* __restrict__ states, const float* __restrict__ acs_g,
             float* __restrict__ pstates) {
  const int b = blockIdx.x / cH;
  const int h = blockIdx.x % cH;
  const int tid = threadIdx.x;
  float run[32];
#pragma unroll
  for (int j = 0; j < 32; ++j) run[j] = 0.f;
  for (int c = 0; c < cNC; ++c) {
    const size_t cidx = (size_t)((b * cNC + c) * cH + h);
    const size_t sidx = cidx * (cP * cN);
    const float dec = __expf(acs_g[cidx * cCH + (cCH - 1)]);
#pragma unroll
    for (int j = 0; j < 32; ++j) {
      const size_t e = sidx + tid + j * 256;
      pstates[e] = run[j];
      run[j] = run[j] * dec + states[e];
    }
  }
}

// ---------------------------------------------------------------------------
// Y_off = (C @ prev_states^T) * exp(acs[l]); y += Y_off + D*hs.
// ---------------------------------------------------------------------------
__global__ void __launch_bounds__(256)
k_yoff(const __bf16* __restrict__ cbf, const float* __restrict__ pstates,
       const float* __restrict__ acs_g, const __bf16* __restrict__ hsb,
       const float* __restrict__ Dp, float* __restrict__ yout) {
  __shared__ __bf16 sC[cCH * cN];   // (l,n)
  __shared__ __bf16 sP[cP * cN];    // (p,n)
  __shared__ float s_acs[cCH];

  const int tid  = threadIdx.x;
  const int wid  = tid >> 5;
  const int lane = tid & 31;
  const int half = lane >> 4;
  const int lr   = lane & 15;

  const int idx = blockIdx.x;
  const int h = idx % cH;
  const int c = (idx / cH) % cNC;
  const int b = idx / (cH * cNC);
  const int g = h % cG;
  const int base = b * cL + c * cCH;

  // async DMA of the C tile; stage sP / s_acs with VALU meanwhile
  for (int ch16 = tid; ch16 < 2048; ch16 += 256) {
    const int l = ch16 >> 4;
    const int n8 = (ch16 & 15) * 8;
    async_ld16(sC + ch16 * 8, cbf + (size_t)(base + l) * cGN + g * cN + n8);
  }
  for (int i = tid; i < cP * cN; i += 256)
    sP[i] = (__bf16)pstates[(size_t)idx * (cP * cN) + i];
  if (tid < cCH) s_acs[tid] = acs_g[(size_t)idx * cCH + tid];
  wait_async0();
  __syncthreads();

  const int lt = wid;
  v8f acc[4] = {};
  const __bf16* arow = sC + (lt * 16 + lr) * cN;
  for (int k0 = 0; k0 < cN; k0 += 32) {
    v16bf fa = frag_a(arow, k0, half);
#pragma unroll
    for (int pt = 0; pt < 4; ++pt) {
      v16bf fb = frag_b(sP + (pt * 16 + lr) * cN, k0, half);
      acc[pt] = wmma_bf16(fa, fb, acc[pt]);
    }
  }
#pragma unroll
  for (int pt = 0; pt < 4; ++pt)
#pragma unroll
    for (int r = 0; r < 8; ++r) {
      const int li = lt * 16 + r + 8 * half;
      const int p = pt * 16 + lr;
      const size_t o = (size_t)(base + li) * cINTER + h * cP + p;
      const float dres = Dp[h] * (float)hsb[o];
      yout[o] += acc[pt][r] * __expf(s_acs[li]) + dres;
    }
}

// ---------------------------------------------------------------------------
// xg = y * silu(gate); per-(row, group of 1024) RMS norm; * norm_weight -> bf16
// ---------------------------------------------------------------------------
__global__ void __launch_bounds__(256)
k_gatenorm(const float* __restrict__ proj, const float* __restrict__ yin,
           const float* __restrict__ norm_w, __bf16* __restrict__ nbf) {
  __shared__ float red[256];
  const int tid = threadIdx.x;
  const int row = blockIdx.x / cG;
  const int g = blockIdx.x % cG;
  constexpr int GW = cINTER / cG;  // 1024

  float v[4];
  float ss = 0.f;
#pragma unroll
  for (int j = 0; j < 4; ++j) {
    const int ch = g * GW + tid + j * 256;
    const float gate = proj[(size_t)row * cPROJ + ch];
    const float xg = yin[(size_t)row * cINTER + ch] * silu_f(gate);
    v[j] = xg;
    ss += xg * xg;
  }
  red[tid] = ss;
  __syncthreads();
  for (int s = 128; s > 0; s >>= 1) {
    if (tid < s) red[tid] += red[tid + s];
    __syncthreads();
  }
  const float rms = rsqrtf(red[0] / (float)GW + cEPS);
#pragma unroll
  for (int j = 0; j < 4; ++j) {
    const int ch = g * GW + tid + j * 256;
    nbf[(size_t)row * cINTER + ch] = (__bf16)(v[j] * rms * norm_w[ch]);
  }
}

// ---------------------------------------------------------------------------
extern "C" void kernel_launch(void* const* d_in, const int* in_sizes, int n_in,
                              void* d_out, int out_size, void* d_ws, size_t ws_size,
                              hipStream_t stream) {
  const float* hid   = (const float*)d_in[0];  // (2, 2048, 4096)
  const float* w_in  = (const float*)d_in[1];  // (18560, 4096)
  const float* cw    = (const float*)d_in[2];  // (10240, 1, 4)
  const float* cb    = (const float*)d_in[3];  // (10240,)
  const float* dtb   = (const float*)d_in[4];  // (128,)
  const float* A_log = (const float*)d_in[5];  // (128,)
  const float* Dp    = (const float*)d_in[6];  // (128,)
  const float* nw    = (const float*)d_in[7];  // (8192,)
  const float* w_out = (const float*)d_in[8];  // (4096, 8192)
  float* out = (float*)d_out;                  // (2, 2048, 4096) f32
  (void)in_sizes; (void)n_in; (void)out_size; (void)ws_size;

  char* ws = (char*)d_ws;
  size_t off = 0;
  auto alloc = [&](size_t bytes) -> char* {
    char* p = ws + off;
    off += (bytes + 255) & ~(size_t)255;
    return p;
  };
  __bf16* hbf  = (__bf16*)alloc((size_t)cBL * cHID * 2);
  __bf16* wibf = (__bf16*)alloc((size_t)cPROJ * cHID * 2);
  __bf16* wobf = (__bf16*)alloc((size_t)cHID * cINTER * 2);
  float*  proj = (float*)alloc((size_t)cBL * cPROJ * 4);
  __bf16* hsb  = (__bf16*)alloc((size_t)cBL * cINTER * 2);
  __bf16* xdt  = (__bf16*)alloc((size_t)cBL * cINTER * 2);
  __bf16* bbf  = (__bf16*)alloc((size_t)cBL * cGN * 2);
  __bf16* cbfp = (__bf16*)alloc((size_t)cBL * cGN * 2);
  float*  acs  = (float*)alloc((size_t)cB * cNC * cH * cCH * 4);
  float*  st   = (float*)alloc((size_t)cB * cNC * cH * cP * cN * 4);
  float*  pst  = (float*)alloc((size_t)cB * cNC * cH * cP * cN * 4);
  float*  y    = (float*)alloc((size_t)cBL * cINTER * 4);
  __bf16* nbf  = (__bf16*)alloc((size_t)cBL * cINTER * 2);

  // 1) casts to bf16
  k_cast_bf16<<<2048, 256, 0, stream>>>(hid, hbf, (size_t)cBL * cHID);
  k_cast_bf16<<<4096, 256, 0, stream>>>(w_in, wibf, (size_t)cPROJ * cHID);
  k_cast_bf16<<<2048, 256, 0, stream>>>(w_out, wobf, (size_t)cHID * cINTER);

  // 2) in_proj: proj = hidden @ in_proj_w^T  (4096 x 18560, K = 4096)
  {
    dim3 grid(cPROJ / 128, cBL / 128);
    k_gemm_nt_bf16<<<grid, 256, 0, stream>>>(hbf, wibf, proj, cBL, cPROJ, cHID);
  }

  // 3) causal conv + SiLU + dt softplus + splits
  {
    const size_t tot = (size_t)cBL * cCONVD;
    k_conv_silu<<<(unsigned)((tot + 255) / 256), 256, 0, stream>>>(
        proj, cw, cb, dtb, hsb, xdt, bbf, cbfp);
  }

  // 4) intra-chunk SSM (WMMA + async LDS DMA), 144 KB dynamic LDS per block
  k_chunk<<<cB * cNC * cH, 256, 73728 * sizeof(__bf16), stream>>>(
      proj, xdt, bbf, cbfp, dtb, A_log, y, st, acs);

  // 5) inter-chunk state recurrence
  k_interchunk<<<cB * cH, 256, 0, stream>>>(st, acs, pst);

  // 6) Y_off + D residual (WMMA + async LDS DMA)
  k_yoff<<<cB * cNC * cH, 256, 0, stream>>>(cbfp, pst, acs, hsb, Dp, y);

  // 7) gate * silu + grouped RMS norm -> bf16
  k_gatenorm<<<cBL * cG, 256, 0, stream>>>(proj, y, nw, nbf);

  // 8) out_proj: out = normed @ out_proj_w^T (4096 x 4096, K = 8192)
  {
    dim3 grid(cHID / 128, cBL / 128);
    k_gemm_nt_bf16<<<grid, 256, 0, stream>>>(nbf, wobf, out, cBL, cHID, cINTER);
  }
}
